// DeformBlockGNN_45165876085120
// MI455X (gfx1250) — compile-verified, matrix operand order checked by reading, not run on hardware
//
#include <hip/hip_runtime.h>
#include <hip/hip_bf16.h>
#include <math.h>

// ---------------------------------------------------------------------------
// Types for CDNA5 WMMA
// ---------------------------------------------------------------------------
typedef __attribute__((ext_vector_type(16))) __bf16 v16bf;
typedef __attribute__((ext_vector_type(8)))  float  v8f;
typedef __attribute__((ext_vector_type(2)))  unsigned u32x2;
typedef __attribute__((ext_vector_type(4)))  unsigned u32x4;

__device__ __forceinline__ unsigned short f32_to_bf16_bits(float f) {
    unsigned u = __float_as_uint(f);
    unsigned r = u + 0x7FFFu + ((u >> 16) & 1u);   // round-to-nearest-even
    return (unsigned short)(r >> 16);
}

__device__ __forceinline__ unsigned pack2bf(float x, float y) {
    return (unsigned)f32_to_bf16_bits(x) | ((unsigned)f32_to_bf16_bits(y) << 16);
}

__device__ __forceinline__ float leaky02(float x) {
    return x > 0.f ? x : 0.2f * x;
}

// ordered-uint mapping so float max == uint max (monotone; 0 = very negative)
__device__ __forceinline__ unsigned f2ord(float f) {
    unsigned u = __float_as_uint(f);
    return (u & 0x80000000u) ? ~u : (u | 0x80000000u);
}
__device__ __forceinline__ float ord2f(unsigned o) {
    unsigned u = (o & 0x80000000u) ? (o & 0x7FFFFFFFu) : ~o;
    return __uint_as_float(u);
}

// ---------------------------------------------------------------------------
// bf16 WMMA GEMM:  C[r, col0+c] = act(A[r,:K] @ B[:K,c] + bias[c])
//   Requirements guaranteed by every call site here:
//     K % 64 == 0, N % 32 == 0, lda % 4 == 0, A/B 16-byte aligned.
//   Block = 128 threads = 4 waves (2x2), tile 32x32, K-step 64 (2 WMMA/iter).
//   LDS: A as [row][k], B transposed as [col][k]; both with stride 72 ushorts
//   (144 B) so each lane's fragment = two contiguous 16-B runs -> ds_load_b128.
// ---------------------------------------------------------------------------
#define KSTEP 64
#define LDSS  72   // ushort row stride (multiple of 8, odd multiple of 4 dwords)

__global__ __launch_bounds__(128)
void wmma_gemm_kernel(const float* __restrict__ A, int lda,
                      const float* __restrict__ B, int ldb,
                      const float* __restrict__ bias,
                      float* __restrict__ C, int ldc, int col0,
                      int rows, int N, int K, int act)
{
    __shared__ unsigned short As[32][LDSS];   // [row][k]
    __shared__ unsigned short Bs[32][LDSS];   // [col][k]  (B stored transposed)

    const int tid  = threadIdx.x;
    const int wave = tid >> 5;
    const int lane = tid & 31;
    const int wr   = wave >> 1;          // wave row (0..1)
    const int wc   = wave & 1;           // wave col (0..1)
    const int half = lane >> 4;          // lane group 0/1
    const int r    = lane & 15;

    const int row0  = blockIdx.x * 32;
    const int colB0 = blockIdx.y * 32;

    // A staging: thread loads float4 at [a_rr + 8*it][a_kk..a_kk+3]
    const int a_rr = tid >> 4;           // 0..7
    const int a_kk = (tid & 15) << 2;    // 0,4,...,60
    // B staging: thread loads 4 k's (b32 each) at column b_cc
    const int b_cc = tid & 31;
    const int b_kk = (tid >> 5) << 2;    // 0,4,8,12

    v8f acc = {0.f, 0.f, 0.f, 0.f, 0.f, 0.f, 0.f, 0.f};

    for (int k0 = 0; k0 < K; k0 += KSTEP) {
        // ---- stage A tile (32 rows x 64 k) as bf16, branchless row clamp ----
#pragma unroll
        for (int it = 0; it < 4; ++it) {
            int rr = a_rr + it * 8;
            int gr = row0 + rr; gr = gr < rows ? gr : rows - 1;
            const float4 a4 = *reinterpret_cast<const float4*>(
                &A[(size_t)gr * lda + (k0 + a_kk)]);
            u32x2 p; p.x = pack2bf(a4.x, a4.y); p.y = pack2bf(a4.z, a4.w);
            *reinterpret_cast<u32x2*>(&As[rr][a_kk]) = p;
        }
        // ---- stage B tile (64 k x 32 cols) transposed into [col][k] ----
#pragma unroll
        for (int it = 0; it < 4; ++it) {
            int kk = b_kk + it * 16;
            const float* pb = &B[(size_t)(k0 + kk) * ldb + (colB0 + b_cc)];
            float b0 = pb[0];
            float b1 = pb[(size_t)ldb];
            float b2 = pb[(size_t)ldb * 2];
            float b3 = pb[(size_t)ldb * 3];
            u32x2 p; p.x = pack2bf(b0, b1); p.y = pack2bf(b2, b3);
            *reinterpret_cast<u32x2*>(&Bs[b_cc][kk]) = p;
        }
        // prefetch next K tile (uniform branch) -> global_prefetch_b8
        if (k0 + KSTEP < K) {
            int gr = row0 + a_rr; gr = gr < rows ? gr : rows - 1;
            __builtin_prefetch(&A[(size_t)gr * lda + (k0 + KSTEP + a_kk)], 0, 3);
            __builtin_prefetch(&B[(size_t)(k0 + KSTEP + b_kk) * ldb + colB0 + b_cc], 0, 3);
        }
        __syncthreads();

        // ---- two WMMAs: fragments are 2x contiguous 16-B LDS reads each ----
#pragma unroll
        for (int j = 0; j < 2; ++j) {
            const int kb = j * 32;
            union { v16bf v; u32x4 q[2]; } af, bfr;
            af.q[0]  = *reinterpret_cast<const u32x4*>(&As[wr * 16 + r][kb + half * 8]);
            af.q[1]  = *reinterpret_cast<const u32x4*>(&As[wr * 16 + r][kb + 16 + half * 8]);
            bfr.q[0] = *reinterpret_cast<const u32x4*>(&Bs[wc * 16 + r][kb + half * 8]);
            bfr.q[1] = *reinterpret_cast<const u32x4*>(&Bs[wc * 16 + r][kb + 16 + half * 8]);
            acc = __builtin_amdgcn_wmma_f32_16x16x32_bf16(
                      false, af.v, false, bfr.v, (short)0, acc, false, false);
        }
        __syncthreads();
    }

    // D layout: VGPR i -> M = i (+8 for lanes 16..31), N = lane&15
#pragma unroll
    for (int i = 0; i < 8; ++i) {
        int row = row0 + wr * 16 + (half ? 8 : 0) + i;
        int col = colB0 + wc * 16 + r;      // always < N (N % 32 == 0)
        if (row < rows) {
            float v = acc[i] + bias[col];
            if (act) v = leaky02(v);
            C[(size_t)row * ldc + col0 + col] = v;
        }
    }
}

// ---------------------------------------------------------------------------
// Trilinear border-clamped sampler: neighbors[m, s] for s in 0..124, pad to 128
// ---------------------------------------------------------------------------
__device__ __forceinline__ float tri_sample(const float* __restrict__ vol,
                                            float px, float py, float pz)
{
    const float Dm1 = 127.f;
    float x = fminf(fmaxf((px + 1.f) * 0.5f * Dm1, 0.f), Dm1);
    float y = fminf(fmaxf((py + 1.f) * 0.5f * Dm1, 0.f), Dm1);
    float z = fminf(fmaxf((pz + 1.f) * 0.5f * Dm1, 0.f), Dm1);
    int x0 = (int)floorf(x), y0 = (int)floorf(y), z0 = (int)floorf(z);
    int x1 = min(x0 + 1, 127), y1 = min(y0 + 1, 127), z1 = min(z0 + 1, 127);
    float fx = x - (float)x0, fy = y - (float)y0, fz = z - (float)z0;
#define VOL(zz, yy, xx) vol[(((size_t)(zz) * 128 + (yy)) * 128 + (xx))]
    float c000 = VOL(z0, y0, x0), c001 = VOL(z0, y0, x1);
    float c010 = VOL(z0, y1, x0), c011 = VOL(z0, y1, x1);
    float c100 = VOL(z1, y0, x0), c101 = VOL(z1, y0, x1);
    float c110 = VOL(z1, y1, x0), c111 = VOL(z1, y1, x1);
#undef VOL
    float c00 = c000 + (c001 - c000) * fx;
    float c01 = c010 + (c011 - c010) * fx;
    float c10 = c100 + (c101 - c100) * fx;
    float c11 = c110 + (c111 - c110) * fx;
    float c0 = c00 + (c01 - c00) * fy;
    float c1 = c10 + (c11 - c10) * fy;
    return c0 + (c1 - c0) * fz;
}

__global__ __launch_bounds__(128)
void sample_kernel(const float* __restrict__ v, const float* __restrict__ vol,
                   float* __restrict__ neigh, int M)
{
    int m = blockIdx.x;
    int s = threadIdx.x;               // 0..127
    if (m >= M) return;
    float val = 0.f;
    if (s < 125) {
        // np.linspace(-3, 2, 5)
        const float g[5] = {-3.f, -1.75f, -0.5f, 0.75f, 2.f};
        int a = s / 25, b = (s / 5) % 5, c = s % 5;
        const float inv = 2.f / 128.f;
        float px = v[m * 3 + 0] + g[a] * inv;
        float py = v[m * 3 + 1] + g[b] * inv;
        float pz = v[m * 3 + 2] + g[c] * inv;
        val = tri_sample(vol, px, py, pz);  // rescale == 1 (cubic volume)
    }
    neigh[(size_t)m * 128 + s] = val;
}

// conv_w [128, 125] -> convB [128(pad K), 128] transposed, zero pad rows
__global__ void conv_transpose_kernel(const float* __restrict__ cw,
                                      float* __restrict__ convB)
{
    int idx = blockIdx.x * blockDim.x + threadIdx.x;
    if (idx >= 128 * 128) return;
    int k = idx >> 7, c = idx & 127;
    convB[idx] = (k < 125) ? cw[c * 125 + k] : 0.f;
}

// ---------------------------------------------------------------------------
// Face normals scatter + normalize
// ---------------------------------------------------------------------------
__global__ void face_normal_kernel(const int* __restrict__ f,
                                   const float* __restrict__ v,
                                   float* __restrict__ vn, int F)
{
    int i = blockIdx.x * blockDim.x + threadIdx.x;
    if (i >= F) return;
    int i0 = f[i * 3 + 0], i1 = f[i * 3 + 1], i2 = f[i * 3 + 2];
    float ax = v[i1 * 3 + 0] - v[i0 * 3 + 0];
    float ay = v[i1 * 3 + 1] - v[i0 * 3 + 1];
    float az = v[i1 * 3 + 2] - v[i0 * 3 + 2];
    float bx = v[i2 * 3 + 0] - v[i0 * 3 + 0];
    float by = v[i2 * 3 + 1] - v[i0 * 3 + 1];
    float bz = v[i2 * 3 + 2] - v[i0 * 3 + 2];
    float nx = ay * bz - az * by;
    float ny = az * bx - ax * bz;
    float nz = ax * by - ay * bx;
    atomicAdd(&vn[i0 * 3 + 0], nx); atomicAdd(&vn[i0 * 3 + 1], ny); atomicAdd(&vn[i0 * 3 + 2], nz);
    atomicAdd(&vn[i1 * 3 + 0], nx); atomicAdd(&vn[i1 * 3 + 1], ny); atomicAdd(&vn[i1 * 3 + 2], nz);
    atomicAdd(&vn[i2 * 3 + 0], nx); atomicAdd(&vn[i2 * 3 + 1], ny); atomicAdd(&vn[i2 * 3 + 2], nz);
}

__global__ void normalize_kernel(float* __restrict__ vn, int M)
{
    int m = blockIdx.x * blockDim.x + threadIdx.x;
    if (m >= M) return;
    float x = vn[m * 3 + 0], y = vn[m * 3 + 1], z = vn[m * 3 + 2];
    float len = fmaxf(sqrtf(x * x + y * y + z * z), 1e-12f);
    vn[m * 3 + 0] = x / len; vn[m * 3 + 1] = y / len; vn[m * 3 + 2] = z / len;
}

// fc1: [v, vn] (6) @ fc1_w [6,128] -> z256[:, 0:128], leaky
__global__ void fc1_kernel(const float* __restrict__ v, const float* __restrict__ vn,
                           const float* __restrict__ w, const float* __restrict__ b,
                           float* __restrict__ z256, int M)
{
    int idx = blockIdx.x * blockDim.x + threadIdx.x;
    if (idx >= M * 128) return;
    int m = idx >> 7, c = idx & 127;
    float x0 = v[m * 3 + 0], x1 = v[m * 3 + 1], x2 = v[m * 3 + 2];
    float x3 = vn[m * 3 + 0], x4 = vn[m * 3 + 1], x5 = vn[m * 3 + 2];
    float s = b[c];
    s += x0 * w[0 * 128 + c] + x1 * w[1 * 128 + c] + x2 * w[2 * 128 + c];
    s += x3 * w[3 * 128 + c] + x4 * w[4 * 128 + c] + x5 * w[5 * 128 + c];
    z256[(size_t)m * 256 + c] = leaky02(s);
}

// edge_attr = concat(t, msg) -> [E, 64]
__global__ void edge_concat_kernel(const float* __restrict__ t,
                                   const float* __restrict__ msg,
                                   float* __restrict__ ea, int E)
{
    long long idx = (long long)blockIdx.x * blockDim.x + threadIdx.x;
    if (idx >= (long long)E * 64) return;
    int e = (int)(idx >> 6), c = (int)(idx & 63);
    ea[idx] = (c < 32) ? t[(size_t)e * 32 + c] : msg[(size_t)e * 32 + (c - 32)];
}

// ---------------------------------------------------------------------------
// Attention: wave-per-edge alpha + segment max (ordered-uint atomicMax)
// ---------------------------------------------------------------------------
__global__ __launch_bounds__(256)
void edge_alpha_kernel(const float* __restrict__ q, const float* __restrict__ kp,
                       const float* __restrict__ ep,
                       const int* __restrict__ src, const int* __restrict__ dst,
                       float* __restrict__ alpha, unsigned* __restrict__ amax, int E)
{
    int wave = threadIdx.x >> 5, lane = threadIdx.x & 31;
    int e = blockIdx.x * 8 + wave;
    if (e >= E) return;
    int s = src[e], d = dst[e];
#pragma unroll
    for (int h = 0; h < 2; ++h) {
        float p = 0.f;
#pragma unroll
        for (int j = 0; j < 2; ++j) {
            int dd = h * 64 + j * 32 + lane;
            float ke = kp[(size_t)s * 128 + dd] + ep[(size_t)e * 128 + dd];
            p += q[(size_t)d * 128 + dd] * ke;
        }
#pragma unroll
        for (int off = 16; off; off >>= 1) p += __shfl_xor(p, off, 32);
        if (lane == 0) {
            float a = p * 0.125f;            // 1/sqrt(64)
            alpha[(size_t)e * 2 + h] = a;
            atomicMax(&amax[d * 2 + h], f2ord(a));
        }
    }
}

// exp(alpha - amax[dst]) in-place, segment-sum denominator
__global__ void edge_exp_kernel(const int* __restrict__ dst, float* __restrict__ alpha,
                                const unsigned* __restrict__ amax,
                                float* __restrict__ den, int E)
{
    long long idx = (long long)blockIdx.x * blockDim.x + threadIdx.x;
    if (idx >= (long long)E * 2) return;
    int e = (int)(idx >> 1), h = (int)(idx & 1);
    int d = dst[e];
    float mx = ord2f(amax[d * 2 + h]);
    float ex = __expf(alpha[idx] - mx);
    alpha[idx] = ex;
    atomicAdd(&den[d * 2 + h], ex);
}

// weighted scatter: out[dst] += w * (v[src] + e_proj)
__global__ __launch_bounds__(256)
void edge_accum_kernel(const int* __restrict__ src, const int* __restrict__ dst,
                       const float* __restrict__ alpha, const float* __restrict__ den,
                       const float* __restrict__ vp, const float* __restrict__ ep,
                       float* __restrict__ out, int E)
{
    int e = blockIdx.x * 2 + (threadIdx.x >> 7);
    if (e >= E) return;
    int dloc = threadIdx.x & 127;
    int h = dloc >> 6;
    int s = src[e], d = dst[e];
    float w = alpha[(size_t)e * 2 + h] / fmaxf(den[d * 2 + h], 1e-16f);
    float ve = vp[(size_t)s * 128 + dloc] + ep[(size_t)e * 128 + dloc];
    atomicAdd(&out[(size_t)d * 128 + dloc], w * ve);
}

// dx = (out @ fco_w [128,3] + fco_b) * 0.1
__global__ void fco_kernel(const float* __restrict__ out, const float* __restrict__ w,
                           const float* __restrict__ b, float* __restrict__ dx, int M)
{
    int idx = blockIdx.x * blockDim.x + threadIdx.x;
    if (idx >= M * 3) return;
    int m = idx / 3, j = idx % 3;
    const float* row = out + (size_t)m * 128;
    float s = b[j];
#pragma unroll 8
    for (int i = 0; i < 128; ++i) s += row[i] * w[i * 3 + j];
    dx[idx] = s * 0.1f;
}

__global__ void zero_kernel(unsigned* __restrict__ p, long long n)
{
    long long i = (long long)blockIdx.x * blockDim.x + threadIdx.x;
    if (i < n) p[i] = 0u;
}

// ---------------------------------------------------------------------------
// Host side
// ---------------------------------------------------------------------------
static inline void launch_gemm(const float* A, int lda, const float* B, int ldb,
                               const float* bias, float* C, int ldc, int col0,
                               int rows, int N, int K, int act, hipStream_t s)
{
    dim3 g((rows + 31) / 32, (N + 31) / 32);
    wmma_gemm_kernel<<<g, 128, 0, s>>>(A, lda, B, ldb, bias, C, ldc, col0,
                                       rows, N, K, act);
}

static inline void launch_zero(void* p, long long n_u32, hipStream_t s)
{
    zero_kernel<<<(unsigned)((n_u32 + 255) / 256), 256, 0, s>>>((unsigned*)p, n_u32);
}

extern "C" void kernel_launch(void* const* d_in, const int* in_sizes, int n_in,
                              void* d_out, int out_size, void* d_ws, size_t ws_size,
                              hipStream_t stream)
{
    // input order: v, f, V, edge_index, t, msg, then params in dict order
    const float* v   = (const float*)d_in[0];
    const int*   f   = (const int*)  d_in[1];
    const float* vol = (const float*)d_in[2];
    const int*   ei  = (const int*)  d_in[3];
    const float* t   = (const float*)d_in[4];
    const float* msg = (const float*)d_in[5];
    const float* fc1_w = (const float*)d_in[6];
    const float* fc1_b = (const float*)d_in[7];
    const float* conv_w = (const float*)d_in[8];
    const float* conv_b = (const float*)d_in[9];
    const float* lfc_w  = (const float*)d_in[10];
    const float* lfc_b  = (const float*)d_in[11];
    const float* fc2_w  = (const float*)d_in[12];
    const float* fc2_b  = (const float*)d_in[13];
    const float* fc3_w  = (const float*)d_in[14];
    const float* fc3_b  = (const float*)d_in[15];
    const float* wq = (const float*)d_in[16];
    const float* bq = (const float*)d_in[17];
    const float* wk = (const float*)d_in[18];
    const float* bk = (const float*)d_in[19];
    const float* wv = (const float*)d_in[20];
    const float* bv = (const float*)d_in[21];
    const float* we = (const float*)d_in[22];
    const float* be = (const float*)d_in[23];
    const float* wskip = (const float*)d_in[24];
    const float* bskip = (const float*)d_in[25];
    const float* fco_w = (const float*)d_in[26];
    const float* fco_b = (const float*)d_in[27];

    const int M = in_sizes[0] / 3;      // 150000
    const int F = in_sizes[1] / 3;      // 300000
    const int E = in_sizes[3] / 2;      // 900000
    const int* src = ei;
    const int* dst = ei + E;

    // workspace layout (f32 units, 64-elem aligned)
    float* base = (float*)d_ws;
    size_t cur = 0;
    auto alloc = [&](size_t n) { float* p = base + cur; cur += (n + 63) & ~(size_t)63; return p; };

    float*    vn     = alloc((size_t)M * 3);
    float*    neigh  = alloc((size_t)M * 128);
    float*    convB  = alloc(128 * 128);
    float*    zl0    = alloc((size_t)M * 128);
    float*    z256   = alloc((size_t)M * 256);
    float*    z512   = alloc((size_t)M * 512);
    float*    z      = alloc((size_t)M * 256);
    float*    qb     = alloc((size_t)M * 128);
    float*    kb     = alloc((size_t)M * 128);
    float*    vb     = alloc((size_t)M * 128);
    float*    outb   = alloc((size_t)M * 128);
    float*    ea     = alloc((size_t)E * 64);
    float*    eproj  = alloc((size_t)E * 128);
    float*    alpha  = alloc((size_t)E * 2);
    unsigned* amax   = (unsigned*)alloc((size_t)M * 2);
    float*    den    = alloc((size_t)M * 2);

    // 0) zero scatter targets
    launch_zero(vn, (long long)M * 3, stream);
    launch_zero(amax, (long long)M * 2, stream);
    launch_zero(den, (long long)M * 2, stream);

    // 1) trilinear neighborhood sampling -> neighbors [M,128] (K padded)
    sample_kernel<<<M, 128, 0, stream>>>(v, vol, neigh, M);

    // 2) conv_w [128,125] -> convB [128,128] (transposed, zero-padded K rows)
    conv_transpose_kernel<<<(128 * 128 + 255) / 256, 256, 0, stream>>>(conv_w, convB);

    // 3) face normals + normalize
    face_normal_kernel<<<(F + 255) / 256, 256, 0, stream>>>(f, v, vn, F);
    normalize_kernel<<<(M + 255) / 256, 256, 0, stream>>>(vn, M);

    // 4) zl = leaky(leaky(neigh @ convB + cb) @ lfc + lb)  -> z256[:,128:256]
    launch_gemm(neigh, 128, convB, 128, conv_b, zl0, 128, 0, M, 128, 128, 1, stream);
    launch_gemm(zl0, 128, lfc_w, 128, lfc_b, z256, 256, 128, M, 128, 128, 1, stream);

    // 5) zp = leaky([v,vn] @ fc1 + b) -> z256[:,0:128]
    fc1_kernel<<<(unsigned)(((long long)M * 128 + 255) / 256), 256, 0, stream>>>(v, vn, fc1_w, fc1_b, z256, M);

    // 6) z = leaky(leaky(z256 @ fc2) @ fc3)
    launch_gemm(z256, 256, fc2_w, 512, fc2_b, z512, 512, 0, M, 512, 256, 1, stream);
    launch_gemm(z512, 512, fc3_w, 256, fc3_b, z, 256, 0, M, 256, 512, 1, stream);

    // 7) q/k/v projections and skip (skip seeds `outb`)
    launch_gemm(z, 256, wq, 128, bq, qb, 128, 0, M, 128, 256, 0, stream);
    launch_gemm(z, 256, wk, 128, bk, kb, 128, 0, M, 128, 256, 0, stream);
    launch_gemm(z, 256, wv, 128, bv, vb, 128, 0, M, 128, 256, 0, stream);
    launch_gemm(z, 256, wskip, 128, bskip, outb, 128, 0, M, 128, 256, 0, stream);

    // 8) edge projection: concat(t,msg)[E,64] @ we[64,128] + be
    edge_concat_kernel<<<(unsigned)(((long long)E * 64 + 255) / 256), 256, 0, stream>>>(t, msg, ea, E);
    launch_gemm(ea, 64, we, 128, be, eproj, 128, 0, E, 128, 64, 0, stream);

    // 9) attention softmax over destination segments
    edge_alpha_kernel<<<(E + 7) / 8, 256, 0, stream>>>(qb, kb, eproj, src, dst, alpha, amax, E);
    edge_exp_kernel<<<(unsigned)(((long long)E * 2 + 255) / 256), 256, 0, stream>>>(dst, alpha, amax, den, E);
    edge_accum_kernel<<<(E + 1) / 2, 256, 0, stream>>>(src, dst, alpha, den, vb, eproj, outb, E);

    // 10) dx = (out @ fco_w + fco_b) * SF
    fco_kernel<<<(unsigned)(((long long)M * 3 + 255) / 256), 256, 0, stream>>>(outb, fco_w, fco_b, (float*)d_out, M);
}
